// SelfKANtentionND_16372415332687
// MI455X (gfx1250) — matrix operand order repaired and checked by reading, not compiled, not run on hardware
//
#include <hip/hip_runtime.h>
#include <hip/hip_bf16.h>

typedef __attribute__((ext_vector_type(16))) _Float16 v16h;
typedef __attribute__((ext_vector_type(8)))  _Float16 v8h;
typedef __attribute__((ext_vector_type(8)))  float    v8f;

#define EPS_F 1e-5f

__device__ inline v16h cat8(v8h a, v8h b) {
  return __builtin_shufflevector(a, b, 0, 1, 2, 3, 4, 5, 6, 7, 8, 9, 10, 11, 12, 13, 14, 15);
}

// ---------------------------------------------------------------------------
// Async global->LDS copy (CDNA5, tracked on ASYNCcnt). Generic LDS pointers
// carry the group-segment offset in the low 32 bits (ISA 10.2 aperture calc),
// which is what VDST of GLOBAL_LOAD_ASYNC_TO_LDS expects.
// ---------------------------------------------------------------------------
__device__ inline void async_copy_b128(void* lds, const void* gaddr) {
  unsigned loff = (unsigned)(unsigned long long)lds;
  asm volatile("global_load_async_to_lds_b128 %0, %1, off"
               :
               : "v"(loff), "v"(gaddr)
               : "memory");
}
__device__ inline void wait_async0() {
  asm volatile("s_wait_asynccnt 0x0" : : : "memory");
}

// ---------------------------------------------------------------------------
// WMMA fragment loaders (wave32 layouts per CDNA5 ISA 7.12.2).
// As : row-major 64x32 f16 tile -> per lane two contiguous 8-half runs.
// BsT: K-major 64x32 tile (BsT[n*32+k]) -> per lane two contiguous 8-half runs (same k order as As).
// C/D: 16x16 f32, 8 VGPRs: row = hi*8 + j, col = lane&15.
// ---------------------------------------------------------------------------
__device__ inline v16h frag_a_lds(const _Float16* As, int row0, int lane) {
  int r = lane & 15, hi = lane >> 4;
  const _Float16* base = As + (row0 + r) * 32;
  v8h lo  = *(const v8h*)(base + hi * 8);
  v8h hi8 = *(const v8h*)(base + 16 + hi * 8);
  return cat8(lo, hi8);
}

__device__ inline v16h frag_b_lds(const _Float16* BsT, int col0, int lane) {
  int n = lane & 15, hi = lane >> 4;
  const _Float16* base = BsT + (col0 + n) * 32;
  v8h lo  = *(const v8h*)(base + hi * 8);
  v8h hi8 = *(const v8h*)(base + 16 + hi * 8);
  return cat8(lo, hi8);
}

// ---------------------------------------------------------------------------
// Epilogue functors
// ---------------------------------------------------------------------------
struct EpF32Bias {   // C f32 = acc + bias[row], row-major
  float* C; long ldc, strideC; const float* bias;
  __device__ void store(float v, int m, int n, int bz) const {
    C[(long)bz * strideC + (long)m * ldc + n] = v + bias[m];
  }
};
struct EpF16Bias {   // C f16 = acc + bias[row], row-major
  _Float16* C; long ldc, strideC; const float* bias;
  __device__ void store(float v, int m, int n, int bz) const {
    C[(long)bz * strideC + (long)m * ldc + n] = (_Float16)(v + bias[m]);
  }
};
struct EpF16BiasT {  // transposed store C[n, m] (K-major producer)
  _Float16* C; long ldc, strideC; const float* bias;
  __device__ void store(float v, int m, int n, int bz) const {
    C[(long)bz * strideC + (long)n * ldc + m] = (_Float16)(v + bias[m]);
  }
};
struct EpF16T {      // transposed store C[n, m], no bias
  _Float16* C; long ldc, strideC;
  __device__ void store(float v, int m, int n, int bz) const {
    C[(long)bz * strideC + (long)n * ldc + m] = (_Float16)v;
  }
};
struct EpF32 {       // raw f32 store (attention scores), row-major
  float* C; long ldc, strideC;
  __device__ void store(float v, int m, int n, int bz) const {
    C[(long)bz * strideC + (long)m * ldc + n] = v;
  }
};
struct EpResid {     // y = acc + b_out[row] + gamma*x  (pre-BN), row-major
  float* Y; const float* X; const float* bias; const float* gamma; long ldc, strideC;
  __device__ void store(float v, int m, int n, int bz) const {
    long idx = (long)bz * strideC + (long)m * ldc + n;
    Y[idx] = v + bias[m] + gamma[0] * X[idx];
  }
};

// ---------------------------------------------------------------------------
// Batched WMMA GEMM: C[M,N] = A(MxK) * B(KxN)
//   A row-major:           (m,k) -> A[m*lda + k]   (lda multiple of 8)
//   B stored K-major [N,K]:(k,n) -> B[n*ldb + k]   (ldb multiple of 8)
// Tiles staged with async global->LDS b128 copies into a double-buffered LDS
// pair (fetch of tile k+1 overlaps WMMA on tile k; ASYNCcnt wait lands after
// compute). All fragments are loaded before the WMMA chain so DS-return
// latency overlaps the matrix pipe instead of gating each WMMA.
// Block = 128 threads (4 waves), 64x64 output tile, K-step 32.
// ---------------------------------------------------------------------------
template <class Ep>
__global__ __launch_bounds__(128)
void gemm_wmma(const _Float16* __restrict__ A, long lda, long strideA,
               const _Float16* __restrict__ B, long ldb, long strideB,
               int M, int N, int K, Ep ep) {
  __shared__ _Float16 As[2][64 * 32];   // row-major [m][k]
  __shared__ _Float16 BsT[2][64 * 32];  // K-major   [n][k]
  const int tid  = threadIdx.x;
  const int lane = tid & 31;
  const int wave = tid >> 5;
  const int n0   = blockIdx.x * 64;
  const int m0   = blockIdx.y * 64;
  const int bz   = blockIdx.z;
  const _Float16* Ab = A + (long)bz * strideA;
  const _Float16* Bb = B + (long)bz * strideB;

  // each thread stages 2 A-chunks + 2 B-chunks (16B each) per K-tile
  const int r0 = tid >> 2;          // 0..31 (chunk row, first half)
  const int c0 = (tid & 3) * 8;     // 0,8,16,24 (halves within row)

  auto stage = [&](int k0, int buf) {
#pragma unroll
    for (int it = 0; it < 2; ++it) {
      int row = r0 + it * 32;       // 0..63
      async_copy_b128(&As[buf][row * 32 + c0],
                      Ab + (long)(m0 + row) * lda + k0 + c0);
      async_copy_b128(&BsT[buf][row * 32 + c0],
                      Bb + (long)(n0 + row) * ldb + k0 + c0);
    }
  };

  v8f zero = {0.f, 0.f, 0.f, 0.f, 0.f, 0.f, 0.f, 0.f};
  v8f acc[4];
#pragma unroll
  for (int t = 0; t < 4; ++t) acc[t] = zero;

  stage(0, 0);
  wait_async0();
  __syncthreads();

  for (int k0 = 0; k0 < K; k0 += 32) {
    const int cur = (k0 >> 5) & 1;
    const bool pre = (k0 + 32) < K;
    if (pre) stage(k0 + 32, cur ^ 1);          // overlaps compute below

    // hoist ALL fragment loads ahead of the WMMA chain: DS returns overlap
    // with the matrix pipe, waits become partial (dscnt > 0) instead of 0.
    v16h a = frag_a_lds(As[cur], wave * 16, lane);
    v16h bfr[4];
#pragma unroll
    for (int nt = 0; nt < 4; ++nt) bfr[nt] = frag_b_lds(BsT[cur], nt * 16, lane);
#pragma unroll
    for (int nt = 0; nt < 4; ++nt)
      acc[nt] = __builtin_amdgcn_wmma_f32_16x16x32_f16(
          false, a, false, bfr[nt], (short)0, acc[nt], false, false);

    if (pre) wait_async0();                    // next tile landed
    __syncthreads();
  }

  const int nlane = lane & 15, hi = lane >> 4;
#pragma unroll
  for (int nt = 0; nt < 4; ++nt)
#pragma unroll
    for (int j = 0; j < 8; ++j)
      ep.store(acc[nt][j], m0 + wave * 16 + hi * 8 + j, n0 + nt * 16 + nlane, bz);
}

// ---------------------------------------------------------------------------
// Elementwise / reduction kernels
// ---------------------------------------------------------------------------
__global__ void f32_to_f16_kernel(const float* __restrict__ in,
                                  _Float16* __restrict__ out, long n) {
  long i = (long)blockIdx.x * blockDim.x + threadIdx.x;
  if (i < n) out[i] = (_Float16)in[i];
}

// x (B,C,N) f32 -> xT (B,N,C) f16  (K-major for the GEMM B operand)
__global__ void x_transpose_f16_kernel(const float* __restrict__ x,
                                       _Float16* __restrict__ xT, int C, int N) {
  long idx = (long)blockIdx.x * blockDim.x + threadIdx.x;  // over B*C*N
  int n = (int)(idx % N);
  long t = idx / N;
  int c = (int)(t % C);
  int b = (int)(t / C);
  xT[((long)b * N + n) * C + c] = (_Float16)x[idx];
}

// Pack [base_w | spline_w] -> W f16 (D x 576), combined bias (D)
__global__ void kan_pack_kernel(const float* __restrict__ base_w,
                                const float* __restrict__ spline_w,
                                const float* __restrict__ base_b,
                                const float* __restrict__ spline_b,
                                _Float16* __restrict__ W, float* __restrict__ bias,
                                int D, int KB, int KS) {
  int idx = blockIdx.x * blockDim.x + threadIdx.x;
  int KT = KB + KS;
  if (idx < D * KT) {
    int o = idx / KT, c = idx % KT;
    float v = (c < KB) ? base_w[o * KB + c] : spline_w[o * KS + (c - KB)];
    W[idx] = (_Float16)v;
  }
  if (idx < D) bias[idx] = base_b[idx] + spline_b[idx];
}

__device__ inline float block_reduce_sum256(float v, float* sh) {
  int tid = threadIdx.x;
  sh[tid] = v; __syncthreads();
  for (int off = 128; off > 0; off >>= 1) {
    if (tid < off) sh[tid] += sh[tid + off];
    __syncthreads();
  }
  float r = sh[0]; __syncthreads();
  return r;
}
__device__ inline float block_reduce_max256(float v, float* sh) {
  int tid = threadIdx.x;
  sh[tid] = v; __syncthreads();
  for (int off = 128; off > 0; off >>= 1) {
    if (tid < off) sh[tid] = fmaxf(sh[tid], sh[tid + off]);
    __syncthreads();
  }
  float r = sh[0]; __syncthreads();
  return r;
}

// instance-norm stats per (b,d) row of att[(B*D), N]
__global__ __launch_bounds__(256)
void inorm_stats_kernel(const float* __restrict__ att, float* __restrict__ mu,
                        float* __restrict__ rs, int N) {
  __shared__ float sh[256];
  int row = blockIdx.x;
  const float* p = att + (long)row * N;
  float s = 0.f, s2 = 0.f;
  for (int i = threadIdx.x; i < N; i += 256) {
    float v = p[i]; s += v; s2 += v * v;
  }
  s  = block_reduce_sum256(s, sh);
  s2 = block_reduce_sum256(s2, sh);
  if (threadIdx.x == 0) {
    float m = s / N;
    float var = s2 / N - m * m;
    mu[row] = m;
    rs[row] = rsqrtf(var + EPS_F);
  }
}

// Build UT (B, N, 576) f16 K-major: col d = silu(att), cols 64+d*8+g = RBF.
// The 8 RBF values per (d) are one contiguous v8h store.
__global__ void build_basis_kernel(const float* __restrict__ att,
                                   const float* __restrict__ mu,
                                   const float* __restrict__ rs,
                                   _Float16* __restrict__ UT,
                                   int D, int N, int KU) {
  long idx = (long)blockIdx.x * blockDim.x + threadIdx.x;  // over B*D*N
  int n = (int)(idx % N);
  long t = idx / N;
  int d = (int)(t % D);
  int b = (int)(t / D);
  float v   = att[idx];
  float sig = 1.f / (1.f + __expf(-v));
  _Float16* row = UT + ((long)b * N + n) * KU;
  row[d] = (_Float16)(v * sig);
  float xn = (v - mu[b * D + d]) * rs[b * D + d];
  const float inv_denom = 7.0f / 4.0f;   // 1 / (4/(G-1))
  v8h basis;
#pragma unroll
  for (int g = 0; g < 8; ++g) {
    float grid = -2.0f + (float)g * (4.0f / 7.0f);
    float z = (xn - grid) * inv_denom;
    basis[g] = (_Float16)__expf(-z * z);
  }
  *(v8h*)(row + D + d * 8) = basis;      // 16B aligned: (64 + 8d)*2 bytes
}

// row softmax of S[N,N] f32 -> P f16 (overwrites S with exp in place)
__global__ __launch_bounds__(256)
void softmax_rows_kernel(float* __restrict__ S, _Float16* __restrict__ P, int N) {
  __shared__ float sh[256];
  int row = blockIdx.x;
  float* s = S + (long)row * N;
  float mx = -3.4e38f;
  for (int i = threadIdx.x; i < N; i += 256) mx = fmaxf(mx, s[i]);
  mx = block_reduce_max256(mx, sh);
  float sum = 0.f;
  for (int i = threadIdx.x; i < N; i += 256) {
    float e = __expf(s[i] - mx);
    s[i] = e;
    sum += e;
  }
  sum = block_reduce_sum256(sum, sh);
  float inv = 1.f / sum;
  _Float16* p = P + (long)row * N;
  for (int i = threadIdx.x; i < N; i += 256) p[i] = (_Float16)(s[i] * inv);
}

// BatchNorm stats over (B, N) per channel c on Y (B, C, N)
__global__ __launch_bounds__(256)
void bn_stats_kernel(const float* __restrict__ Y, float* __restrict__ mean,
                     float* __restrict__ rstd, int B, int C, int N) {
  __shared__ float sh[256];
  int c = blockIdx.x;
  float s = 0.f, s2 = 0.f;
  for (int b = 0; b < B; ++b) {
    const float* p = Y + ((long)b * C + c) * N;
    for (int i = threadIdx.x; i < N; i += 256) {
      float v = p[i]; s += v; s2 += v * v;
    }
  }
  s  = block_reduce_sum256(s, sh);
  s2 = block_reduce_sum256(s2, sh);
  if (threadIdx.x == 0) {
    float cnt = (float)B * (float)N;
    float m = s / cnt;
    float var = s2 / cnt - m * m;
    mean[c] = m;
    rstd[c] = rsqrtf(var + EPS_F);
  }
}

__global__ void bn_apply_kernel(const float* __restrict__ Y,
                                const float* __restrict__ mean,
                                const float* __restrict__ rstd,
                                const float* __restrict__ w,
                                const float* __restrict__ bcoef,
                                float* __restrict__ out, int C, int N) {
  long idx = (long)blockIdx.x * blockDim.x + threadIdx.x;
  int c = (int)((idx / N) % C);
  out[idx] = w[c] * (Y[idx] - mean[c]) * rstd[c] + bcoef[c];
}

// ---------------------------------------------------------------------------
// Host launcher
// ---------------------------------------------------------------------------
extern "C" void kernel_launch(void* const* d_in, const int* in_sizes, int n_in,
                              void* d_out, int out_size, void* d_ws, size_t ws_size,
                              hipStream_t stream) {
  (void)in_sizes; (void)n_in; (void)out_size; (void)ws_size;
  const int  Bn = 4, Cin = 256, Dh = 64, N = 4096, KU = 576;

  const float* x          = (const float*)d_in[0];
  const float* w_in       = (const float*)d_in[1];
  const float* b_in       = (const float*)d_in[2];
  const float* q_base_w   = (const float*)d_in[3];
  const float* q_base_b   = (const float*)d_in[4];
  const float* q_spline_w = (const float*)d_in[5];
  const float* q_spline_b = (const float*)d_in[6];
  const float* k_base_w   = (const float*)d_in[7];
  const float* k_base_b   = (const float*)d_in[8];
  const float* k_spline_w = (const float*)d_in[9];
  const float* k_spline_b = (const float*)d_in[10];
  const float* v_base_w   = (const float*)d_in[11];
  const float* v_base_b   = (const float*)d_in[12];
  const float* v_spline_w = (const float*)d_in[13];
  const float* v_spline_b = (const float*)d_in[14];
  const float* w_out      = (const float*)d_in[15];
  const float* b_out      = (const float*)d_in[16];
  const float* gamma      = (const float*)d_in[17];
  const float* bn_w       = (const float*)d_in[18];
  const float* bn_b       = (const float*)d_in[19];
  float* out = (float*)d_out;

  // ---- workspace carve-up (256B aligned) ----
  char* ws = (char*)d_ws;
  size_t off = 0;
  auto take = [&](size_t bytes) -> void* {
    void* p = ws + off;
    off += (bytes + 255) & ~(size_t)255;
    return p;
  };
  _Float16* xT    = (_Float16*)take((size_t)Bn * N * Cin * 2);   // [b][n][c]
  float*    att   = (float*)   take((size_t)Bn * Dh * N * 4);    // [b][d][n]
  float*    mu    = (float*)   take((size_t)Bn * Dh * 4);
  float*    rs    = (float*)   take((size_t)Bn * Dh * 4);
  _Float16* UT    = (_Float16*)take((size_t)Bn * N * KU * 2);    // [b][n][576]
  _Float16* Wq    = (_Float16*)take((size_t)Dh * KU * 2);
  _Float16* Wk    = (_Float16*)take((size_t)Dh * KU * 2);
  _Float16* Wv    = (_Float16*)take((size_t)Dh * KU * 2);
  float*    qb    = (float*)   take(Dh * 4);
  float*    kb    = (float*)   take(Dh * 4);
  float*    vb    = (float*)   take(Dh * 4);
  _Float16* w_inh  = (_Float16*)take((size_t)Dh * Cin * 2);
  _Float16* w_outh = (_Float16*)take((size_t)Cin * Dh * 2);
  _Float16* qT    = (_Float16*)take((size_t)Bn * N * Dh * 2);    // [b][n][d]
  _Float16* kT    = (_Float16*)take((size_t)Bn * N * Dh * 2);    // [b][n][d]
  _Float16* vh    = (_Float16*)take((size_t)Bn * Dh * N * 2);    // [b][d][n]
  float*    S     = (float*)   take((size_t)N * N * 4);          // per-batch
  _Float16* P     = (_Float16*)take((size_t)N * N * 2);          // per-batch
  _Float16* OT    = (_Float16*)take((size_t)Bn * N * Dh * 2);    // [b][n][d]
  float*    Y     = (float*)   take((size_t)Bn * Cin * N * 4);
  float*    bnm   = (float*)   take(Cin * 4);
  float*    bnr   = (float*)   take(Cin * 4);

  // ---- 1. precision prep: x -> K-major f16, weights -> f16, KAN packing ----
  {
    long n = (long)Bn * Cin * N;
    x_transpose_f16_kernel<<<(int)((n + 255) / 256), 256, 0, stream>>>(x, xT, Cin, N);
    long nw = (long)Dh * Cin;
    f32_to_f16_kernel<<<(int)((nw + 255) / 256), 256, 0, stream>>>(w_in, w_inh, nw);
    f32_to_f16_kernel<<<(int)((nw + 255) / 256), 256, 0, stream>>>(w_out, w_outh, nw);
    int blk = (Dh * KU + 255) / 256;
    kan_pack_kernel<<<blk, 256, 0, stream>>>(q_base_w, q_spline_w, q_base_b, q_spline_b, Wq, qb, Dh, Dh, Dh * 8);
    kan_pack_kernel<<<blk, 256, 0, stream>>>(k_base_w, k_spline_w, k_base_b, k_spline_b, Wk, kb, Dh, Dh, Dh * 8);
    kan_pack_kernel<<<blk, 256, 0, stream>>>(v_base_w, v_spline_w, v_base_b, v_spline_b, Wv, vb, Dh, Dh, Dh * 8);
  }

  // ---- 2. inner projection: att[b] = w_in (64x256) * x[b] (256x4096) + b_in ----
  {
    EpF32Bias ep{att, N, (long)Dh * N, b_in};
    gemm_wmma<EpF32Bias><<<dim3(N / 64, 1, Bn), 128, 0, stream>>>(
        w_inh, Cin, 0, xT, Cin, (long)N * Cin, Dh, N, Cin, ep);
  }

  // ---- 3. instance-norm stats + fused silu/RBF basis matrix UT ----
  inorm_stats_kernel<<<Bn * Dh, 256, 0, stream>>>(att, mu, rs, N);
  {
    long n = (long)Bn * Dh * N;
    build_basis_kernel<<<(int)(n / 256), 256, 0, stream>>>(att, mu, rs, UT, Dh, N, KU);
  }

  // ---- 4. q/k/v: [base|spline] (64x576) * U[b] (576x4096) + combined bias ----
  //      q,k written K-major ([n][d]) for the score GEMM; v stays [d][n].
  {
    EpF16BiasT epq{qT, Dh, (long)N * Dh, qb};
    EpF16BiasT epk{kT, Dh, (long)N * Dh, kb};
    EpF16Bias  epv{vh, N, (long)Dh * N, vb};
    gemm_wmma<EpF16BiasT><<<dim3(N / 64, 1, Bn), 128, 0, stream>>>(
        Wq, KU, 0, UT, KU, (long)N * KU, Dh, N, KU, epq);
    gemm_wmma<EpF16BiasT><<<dim3(N / 64, 1, Bn), 128, 0, stream>>>(
        Wk, KU, 0, UT, KU, (long)N * KU, Dh, N, KU, epk);
    gemm_wmma<EpF16Bias><<<dim3(N / 64, 1, Bn), 128, 0, stream>>>(
        Wv, KU, 0, UT, KU, (long)N * KU, Dh, N, KU, epv);
  }

  // ---- 5. attention, batch-sequential so S/P stay resident in L2 ----
  for (int b = 0; b < Bn; ++b) {
    const _Float16* qT_ = qT + (long)b * N * Dh;
    const _Float16* kT_ = kT + (long)b * N * Dh;
    const _Float16* vh_ = vh + (long)b * Dh * N;
    // S[nq,nk] = sum_d qT[nq,d] * kT[nk,d]
    EpF32 eps{S, N, 0};
    gemm_wmma<EpF32><<<dim3(N / 64, N / 64, 1), 128, 0, stream>>>(
        qT_, Dh, 0, kT_, Dh, 0, N, N, Dh, eps);
    softmax_rows_kernel<<<N, 256, 0, stream>>>(S, P, N);
    // O[d,nq] = sum_m v[d,m] * P[nq,m]; P rows are K-contiguous already.
    EpF16T epo{OT + (long)b * N * Dh, Dh, 0};
    gemm_wmma<EpF16T><<<dim3(N / 64, 1, 1), 128, 0, stream>>>(
        vh_, N, 0, P, N, 0, Dh, N, N, epo);
  }

  // ---- 6. outer projection + residual: Y = w_out*O + b_out + gamma*x ----
  {
    EpResid ep{Y, x, b_out, gamma, N, (long)Cin * N};
    gemm_wmma<EpResid><<<dim3(N / 64, Cin / 64, Bn), 128, 0, stream>>>(
        w_outh, Dh, 0, OT, Dh, (long)N * Dh, Cin, N, Dh, ep);
  }

  // ---- 7. training-mode BatchNorm over (B,H,W) ----
  bn_stats_kernel<<<Cin, 256, 0, stream>>>(Y, bnm, bnr, Bn, Cin, N);
  {
    long n = (long)Bn * Cin * N;
    bn_apply_kernel<<<(int)(n / 256), 256, 0, stream>>>(Y, bnm, bnr, bn_w, bn_b, out, Cin, N);
  }
}
